// WindowAttention_64682207478498
// MI455X (gfx1250) — matrix operand order repaired and checked by reading, not compiled, not run on hardware
//
#include <hip/hip_runtime.h>
#include <hip/hip_bf16.h>

typedef _Float16 v16h __attribute__((ext_vector_type(16)));
typedef _Float16 v8h  __attribute__((ext_vector_type(8)));
typedef float    v8f  __attribute__((ext_vector_type(8)));
typedef unsigned int u32x4 __attribute__((ext_vector_type(4)));
typedef int          i32x8 __attribute__((ext_vector_type(8)));
typedef int          i32x4 __attribute__((ext_vector_type(4)));

#define NUM_HEADS 16
#define NTOK      49          // tokens per window
#define NPAD      64          // padded tokens
#define DIM       512
#define QKVDIM    1536
#define HDIM      32          // head dim
#define BWIN      2048        // windows * batch
#define NW        64          // distinct masks
#define MTOT      (BWIN * NTOK)            // 100352 rows
#define QKV_E     ((size_t)MTOT * QKVDIM)  // halfs in fused qkv matrix

#define LDA 40   // LDS stride (halfs) for 32-wide K tiles (+8 pad)
#define LDP 72   // LDS stride (halfs) for 64-wide tiles (+8 pad)
#define LDS_S 68 // LDS stride (floats) for the 64-wide score tile
#define LDC 144  // LDS stride (halfs) for 128-wide epilogue tile (+16 pad)

__device__ __forceinline__ v16h mk16(const _Float16* lo, const _Float16* hi8) {
    v8h a = *(const v8h*)lo;
    v8h b = *(const v8h*)hi8;
    v16h r;
#pragma unroll
    for (int i = 0; i < 8; ++i) { r[i] = a[i]; r[i + 8] = b[i]; }
    return r;
}

// ---------------------------------------------------------------------------
// TDM: DMA one 2D tile (NTOK rows x HDIM halfs, row stride QKVDIM halfs in
// global) into LDS with a 40-half destination row stride via TDM padding
// (pad 4 dwords after every 16 dwords stored).  D# layout per ISA 08 §8.
// This toolchain exposes the 6-arg builtin (g0, g1, g2, g3, g4, cpol).
// ---------------------------------------------------------------------------
__device__ __forceinline__ void tdm_load_tile_2d(void* lds_dst, const void* gsrc) {
    unsigned long long ga = (unsigned long long)gsrc;
    unsigned lds_off = (unsigned)(unsigned long long)lds_dst; // low 32b of LDS aperture = LDS byte offset
    u32x4 g0;
    g0[0] = 1u;                                   // count=1, no gather
    g0[1] = lds_off;                              // lds_addr
    g0[2] = (unsigned)ga;                         // global_addr[31:0]
    g0[3] = (unsigned)((ga >> 32) & 0x1FFFFFFull) // global_addr[56:32]
          | (2u << 30);                           // type = 2 (image)
    i32x8 g1;
    g1[0] = (1 << 16)        // data_size = 1 -> 2 bytes
          | (1 << 20)        // pad_enable
          | (3 << 22)        // pad_interval: 16 dwords (one 32-half row)
          | (3 << 25);       // pad_amount: 4 dwords (8 halfs) -> 40-half stride
    g1[1] = (HDIM & 0xFFFF) << 16;   // tensor_dim0 = 32 (bits 79:48, low half)
    g1[2] = (NTOK << 16);            // tensor_dim0 hi | tensor_dim1 = 49 (low)
    g1[3] = (HDIM << 16);            // tensor_dim1 hi | tile_dim0 = 32
    g1[4] = NTOK;                    // tile_dim1 = 49, tile_dim2 = 0
    g1[5] = QKVDIM;                  // tensor_dim0_stride = 1536 (low 32)
    g1[6] = 0;                       // stride0 hi | stride1 lo
    g1[7] = 0;
    i32x4 z4 = {0, 0, 0, 0};
    i32x8 z8 = {0, 0, 0, 0, 0, 0, 0, 0};
    __builtin_amdgcn_tensor_load_to_lds(g0, g1, z4, z4, z8, 0);
}

// ---------------------------------------------------------------------------
// Kernel 1: fused QKV GEMM.  qkv[100352 x 1536] f16 = x @ qkv_w + qkv_b.
// 128x128 block tile, 8 waves x (64x32).  Epilogue restages the output tile
// through LDS for coalesced 16B stores.
// ---------------------------------------------------------------------------
__global__ __launch_bounds__(256)
void qkv_gemm_kernel(const float* __restrict__ x,
                     const float* __restrict__ w,      // [512,1536]
                     const float* __restrict__ bias,   // [1536]
                     _Float16* __restrict__ qkv) {     // [M,1536] f16
    __shared__ __align__(16) _Float16 smem[128 * LDC]; // 36864 B, reused
    _Float16* As = smem;               // 128*LDA, row-major M x K
    _Float16* Bs = smem + 128 * LDA;   // 128*LDA, transposed N x K
    _Float16* Cs = smem;               // epilogue: 128 x LDC

    const int tid   = threadIdx.x;
    const int lane  = tid & 31;
    const int lm    = lane & 15;
    const int lhalf = lane >> 4;
    const int wid   = tid >> 5;
    const int wm    = wid >> 2;            // 0..1
    const int wn    = wid & 3;             // 0..3

    const int m0 = blockIdx.x * 128;
    const int n0 = blockIdx.y * 128;

    v8f acc[4][2];
    v8f zero = {};
#pragma unroll
    for (int i = 0; i < 4; ++i)
#pragma unroll
        for (int j = 0; j < 2; ++j) acc[i][j] = zero;

    for (int kk = 0; kk < DIM; kk += 32) {
#pragma unroll
        for (int i = 0; i < 4; ++i) {          // A: 128x32 f32 -> f16
            int chunk = tid + i * 256;
            int r  = chunk >> 3;
            int c4 = chunk & 7;
            float4 xv = *(const float4*)&x[(size_t)(m0 + r) * DIM + kk + c4 * 4];
            _Float16* d = &As[r * LDA + c4 * 4];
            d[0] = (_Float16)xv.x; d[1] = (_Float16)xv.y;
            d[2] = (_Float16)xv.z; d[3] = (_Float16)xv.w;
        }
#pragma unroll
        for (int i = 0; i < 4; ++i) {          // B transposed: Bs[n][k]
            int chunk = tid + i * 256;
            int kr  = chunk >> 5;
            int nc4 = chunk & 31;
            float4 wv = *(const float4*)&w[(size_t)(kk + kr) * QKVDIM + n0 + nc4 * 4];
            Bs[(nc4 * 4 + 0) * LDA + kr] = (_Float16)wv.x;
            Bs[(nc4 * 4 + 1) * LDA + kr] = (_Float16)wv.y;
            Bs[(nc4 * 4 + 2) * LDA + kr] = (_Float16)wv.z;
            Bs[(nc4 * 4 + 3) * LDA + kr] = (_Float16)wv.w;
        }
        __syncthreads();

        v16h bfrag[2];
#pragma unroll
        for (int nt = 0; nt < 2; ++nt) {
            int col = wn * 32 + nt * 16 + lm;
            int kb  = lhalf * 8;
            bfrag[nt] = mk16(&Bs[col * LDA + kb], &Bs[col * LDA + kb + 16]);
        }
#pragma unroll
        for (int mt = 0; mt < 4; ++mt) {
            int row = wm * 64 + mt * 16 + lm;
            int kb  = lhalf * 8;
            v16h a = mk16(&As[row * LDA + kb], &As[row * LDA + kb + 16]);
#pragma unroll
            for (int nt = 0; nt < 2; ++nt)
                acc[mt][nt] = __builtin_amdgcn_wmma_f32_16x16x32_f16(
                    false, a, false, bfrag[nt], (short)0, acc[mt][nt], false, false);
        }
        __syncthreads();
    }

    // epilogue: bias add into LDS tile, then coalesced b128 stores
#pragma unroll
    for (int mt = 0; mt < 4; ++mt)
#pragma unroll
        for (int nt = 0; nt < 2; ++nt) {
            v8f c = acc[mt][nt];
            int col = wn * 32 + nt * 16 + lm;
            float bv = bias[n0 + col];
#pragma unroll
            for (int r = 0; r < 8; ++r) {
                int row = wm * 64 + mt * 16 + r + lhalf * 8;
                Cs[row * LDC + col] = (_Float16)(c[r] + bv);
            }
        }
    __syncthreads();
#pragma unroll
    for (int i = 0; i < 8; ++i) {
        int chunk = tid + i * 256;            // 128 rows x 16 chunks
        int r   = chunk >> 4;
        int off = (chunk & 15) * 8;
        v8h val = *(const v8h*)&Cs[r * LDC + off];
        *(v8h*)&qkv[(size_t)(m0 + r) * QKVDIM + n0 + off] = val;
    }
}

// ---------------------------------------------------------------------------
// Kernel 2: per-(window, head) attention.  Q/K staged by the Tensor Data
// Mover (strided 49x32 tiles out of the fused qkv matrix), V staged with
// coalesced vector loads + LDS transpose.
// ---------------------------------------------------------------------------
__global__ __launch_bounds__(128)
void attn_kernel(const _Float16* __restrict__ qkv,     // [M,1536] f16
                 const float* __restrict__ bias_table, // [169,16]
                 const int*   __restrict__ rel_idx,    // [2401]
                 const float* __restrict__ mask,       // [64,49,49]
                 _Float16* __restrict__ out) {         // [M,512] f16
    __shared__ __align__(16) _Float16 Qs[NPAD * LDA];
    __shared__ __align__(16) _Float16 Ks[NPAD * LDA];
    __shared__ __align__(16) _Float16 Vt[HDIM * LDP];   // transposed [d][m]
    __shared__ __align__(16) float    Ss[NPAD * LDS_S];
    __shared__ __align__(16) _Float16 Ps[NPAD * LDP];

    const int b = blockIdx.x;
    const int h = blockIdx.y;
    const int tid   = threadIdx.x;
    const int lane  = tid & 31;
    const int lm    = lane & 15;
    const int lhalf = lane >> 4;
    const int w     = tid >> 5;

    const _Float16* qrow = qkv + (size_t)(b * NTOK) * QKVDIM + h * HDIM;
    const _Float16* krow = qrow + DIM;
    const _Float16* vrow = qrow + 2 * DIM;

    // TDM: DMA Q and K tiles into LDS (wave 0 / wave 1 issue; EXEC ignored,
    // branch is wave-uniform).  Pad rows 49..63 and transposed V by VALU.
    if (w == 0) tdm_load_tile_2d(&Qs[0], qrow);
    if (w == 1) tdm_load_tile_2d(&Ks[0], krow);
    for (int idx = tid; idx < NPAD * HDIM; idx += 128) {
        int m = idx >> 5, d = idx & 31;
        Vt[d * LDP + m] = (m < NTOK) ? vrow[(size_t)m * QKVDIM + d] : (_Float16)0.0f;
        if (m >= NTOK) {
            Qs[m * LDA + d] = (_Float16)0.0f;
            Ks[m * LDA + d] = (_Float16)0.0f;
        }
    }
    __builtin_amdgcn_s_wait_tensorcnt(0);
    __syncthreads();

    // ---- S = Q K^T  (K = 32 -> one WMMA per 16x16 tile) ----
    {
        int row = w * 16 + lm;
        int kb  = lhalf * 8;
        v16h a = mk16(&Qs[row * LDA + kb], &Qs[row * LDA + kb + 16]);
        v8f sacc[4];
        v8f zero = {};
#pragma unroll
        for (int nt = 0; nt < 4; ++nt) {
            int col = nt * 16 + lm;
            int kbb = lhalf * 8;
            v16h bf = mk16(&Ks[col * LDA + kbb], &Ks[col * LDA + kbb + 16]);
            sacc[nt] = __builtin_amdgcn_wmma_f32_16x16x32_f16(
                false, a, false, bf, (short)0, zero, false, false);
        }
        const float scale = 0.17677669529663687f;  // 32^-0.5
        const float* mrow = mask + (size_t)(b & (NW - 1)) * NTOK * NTOK;
#pragma unroll
        for (int nt = 0; nt < 4; ++nt)
#pragma unroll
            for (int r = 0; r < 8; ++r) {
                int n = w * 16 + r + lhalf * 8;
                int m = nt * 16 + lm;
                float o;
                if (n >= NTOK)       o = 0.0f;     // pad row: discarded later
                else if (m >= NTOK)  o = -1e30f;   // pad col: exp -> 0
                else o = sacc[nt][r] * scale
                       + bias_table[rel_idx[n * NTOK + m] * NUM_HEADS + h]
                       + mrow[n * NTOK + m];
                Ss[n * LDS_S + m] = o;
            }
    }
    __syncthreads();

    // ---- fp32 softmax per row ----
    if (tid < NPAD) {
        int row = tid;
        float mx = -3.4e38f;
        for (int j = 0; j < NPAD; ++j) mx = fmaxf(mx, Ss[row * LDS_S + j]);
        float s = 0.0f;
        for (int j = 0; j < NPAD; ++j) {
            float e = __expf(Ss[row * LDS_S + j] - mx);
            Ss[row * LDS_S + j] = e;
            s += e;
        }
        float rinv = 1.0f / s;
        for (int j = 0; j < NPAD; ++j)
            Ps[row * LDP + j] = (_Float16)(Ss[row * LDS_S + j] * rinv);
    }
    __syncthreads();

    // ---- O = P @ V  (2 K-steps x 2 N-tiles) ----
    {
        v8f oacc[2];
        v8f zero = {};
        oacc[0] = zero; oacc[1] = zero;
#pragma unroll
        for (int ks = 0; ks < 2; ++ks) {
            int row = w * 16 + lm;
            int kb  = ks * 32 + lhalf * 8;
            v16h a = mk16(&Ps[row * LDP + kb], &Ps[row * LDP + kb + 16]);
#pragma unroll
            for (int nt = 0; nt < 2; ++nt) {
                int dcol = nt * 16 + lm;
                int kr   = ks * 32 + lhalf * 8;
                v16h bf = mk16(&Vt[dcol * LDP + kr], &Vt[dcol * LDP + kr + 16]);
                oacc[nt] = __builtin_amdgcn_wmma_f32_16x16x32_f16(
                    false, a, false, bf, (short)0, oacc[nt], false, false);
            }
        }
#pragma unroll
        for (int nt = 0; nt < 2; ++nt)
#pragma unroll
            for (int r = 0; r < 8; ++r) {
                int n = w * 16 + r + lhalf * 8;
                if (n < NTOK) {
                    int d = nt * 16 + lm;
                    out[(size_t)(b * NTOK + n) * DIM + h * HDIM + d] =
                        (_Float16)oacc[nt][r];
                }
            }
    }
}

// ---------------------------------------------------------------------------
// Kernel 3: output projection.  out[100352 x 512] f32 = A_f16 @ proj_w + b.
// ---------------------------------------------------------------------------
__global__ __launch_bounds__(256)
void proj_gemm_kernel(const _Float16* __restrict__ a16,  // [M,512] f16
                      const float* __restrict__ w,       // [512,512]
                      const float* __restrict__ bias,    // [512]
                      float* __restrict__ out) {         // [M,512] f32
    __shared__ __align__(16) _Float16 As[128 * LDA];
    __shared__ __align__(16) _Float16 Bs[128 * LDA];

    const int tid   = threadIdx.x;
    const int lane  = tid & 31;
    const int lm    = lane & 15;
    const int lhalf = lane >> 4;
    const int wid   = tid >> 5;
    const int wm    = wid >> 2;
    const int wn    = wid & 3;

    const int m0 = blockIdx.x * 128;
    const int n0 = blockIdx.y * 128;

    v8f acc[4][2];
    v8f zero = {};
#pragma unroll
    for (int i = 0; i < 4; ++i)
#pragma unroll
        for (int j = 0; j < 2; ++j) acc[i][j] = zero;

    for (int kk = 0; kk < DIM; kk += 32) {
#pragma unroll
        for (int i = 0; i < 2; ++i) {          // A: 128x32 halfs, b128 chunks
            int chunk = tid + i * 256;
            int r   = chunk >> 2;
            int off = (chunk & 3) * 8;
            v8h av = *(const v8h*)&a16[(size_t)(m0 + r) * DIM + kk + off];
            *(v8h*)&As[r * LDA + off] = av;
        }
#pragma unroll
        for (int i = 0; i < 4; ++i) {          // B transposed: Bs[n][k]
            int chunk = tid + i * 256;
            int kr  = chunk >> 5;
            int nc4 = chunk & 31;
            float4 wv = *(const float4*)&w[(size_t)(kk + kr) * DIM + n0 + nc4 * 4];
            Bs[(nc4 * 4 + 0) * LDA + kr] = (_Float16)wv.x;
            Bs[(nc4 * 4 + 1) * LDA + kr] = (_Float16)wv.y;
            Bs[(nc4 * 4 + 2) * LDA + kr] = (_Float16)wv.z;
            Bs[(nc4 * 4 + 3) * LDA + kr] = (_Float16)wv.w;
        }
        __syncthreads();

        v16h bfrag[2];
#pragma unroll
        for (int nt = 0; nt < 2; ++nt) {
            int col = wn * 32 + nt * 16 + lm;
            int kb  = lhalf * 8;
            bfrag[nt] = mk16(&Bs[col * LDA + kb], &Bs[col * LDA + kb + 16]);
        }
#pragma unroll
        for (int mt = 0; mt < 4; ++mt) {
            int row = wm * 64 + mt * 16 + lm;
            int kb  = lhalf * 8;
            v16h a = mk16(&As[row * LDA + kb], &As[row * LDA + kb + 16]);
#pragma unroll
            for (int nt = 0; nt < 2; ++nt)
                acc[mt][nt] = __builtin_amdgcn_wmma_f32_16x16x32_f16(
                    false, a, false, bfrag[nt], (short)0, acc[mt][nt], false, false);
        }
        __syncthreads();
    }

    // direct f32 stores: 16 lanes cover 64B contiguous segments per row
#pragma unroll
    for (int mt = 0; mt < 4; ++mt)
#pragma unroll
        for (int nt = 0; nt < 2; ++nt) {
            v8f c = acc[mt][nt];
            int gn = n0 + wn * 32 + nt * 16 + lm;
            float bv = bias[gn];
#pragma unroll
            for (int r = 0; r < 8; ++r) {
                int gm = m0 + wm * 64 + mt * 16 + r + lhalf * 8;
                out[(size_t)gm * DIM + gn] = c[r] + bv;
            }
        }
}

// ---------------------------------------------------------------------------
extern "C" void kernel_launch(void* const* d_in, const int* in_sizes, int n_in,
                              void* d_out, int out_size, void* d_ws, size_t ws_size,
                              hipStream_t stream) {
    const float* x          = (const float*)d_in[0];
    const float* mask       = (const float*)d_in[1];
    const float* qkv_w      = (const float*)d_in[2];
    const float* qkv_b      = (const float*)d_in[3];
    const float* proj_w     = (const float*)d_in[4];
    const float* proj_b     = (const float*)d_in[5];
    const float* bias_table = (const float*)d_in[6];
    const int*   rel_idx    = (const int*)d_in[7];
    float* out = (float*)d_out;

    _Float16* qkv_ws = (_Float16*)d_ws;                 // [M,1536] f16
    _Float16* attn_ws = qkv_ws + QKV_E;                 // [M,512]  f16

    qkv_gemm_kernel<<<dim3(MTOT / 128, QKVDIM / 128), 256, 0, stream>>>(
        x, qkv_w, qkv_b, qkv_ws);

    attn_kernel<<<dim3(BWIN, NUM_HEADS), 128, 0, stream>>>(
        qkv_ws, bias_table, rel_idx, mask, attn_ws);

    proj_gemm_kernel<<<dim3(MTOT / 128, DIM / 128), 256, 0, stream>>>(
        attn_ws, proj_w, proj_b, out);
}
